// NeuronGptOssDecoderLayer_4956392260111
// MI455X (gfx1250) — compile-verified, matrix operand order checked by reading, not run on hardware
//
#include <hip/hip_runtime.h>
#include <hip/hip_bf16.h>
#include <cstdint>

// ---------------------------------------------------------------------------
// Types for CDNA5 WMMA (wave32): A/B = 16 x bf16 (8 VGPRs), C/D = 8 x f32.
// ---------------------------------------------------------------------------
typedef __attribute__((ext_vector_type(16))) __bf16 v16bf;
typedef __attribute__((ext_vector_type(8)))  float  v8f;
typedef __attribute__((ext_vector_type(2)))  float  v2f;
typedef __attribute__((ext_vector_type(2)))  __bf16 v2bf;

union Frag {
    v16bf v;
    uint4 q[2];
    unsigned short s[16];
};

// Packed f32x2 -> bf16x2 (lets the backend select v_cvt_pk_bf16_f32).
__device__ __forceinline__ unsigned cvt2(float x, float y) {
    v2f f; f[0] = x; f[1] = y;
    v2bf b = __builtin_convertvector(f, v2bf);
    union { v2bf b; unsigned u; } u2; u2.b = b;
    return u2.u;
}
__device__ __forceinline__ unsigned short cvt1(float x) {
    union { __bf16 b; unsigned short s; } u1; u1.b = (__bf16)x;
    return u1.s;
}

// ---------------------------------------------------------------------------
// Generic GEMM:  C[M,N] (+)= A[M,K] * B
//   BT=true : B is [N,K] row-major (X @ W^T);  BT=false: B is [K,N] row-major
//   EPI=0: C = acc
//   EPI=1: C = acc + aux[m*N+n]                  (residual add)
//   EPI=2: C = C + aux[tok*auxStride] * acc      (MoE routed scatter-accum)
//   EPI=3: C = silu(acc) * aux[m*N+n]            (fused SwiGLU gate)
//   GM=0: direct. GM=1: A rows gathered via rowIdx, compact C, early-exit
//   past *cntPtr. GM=2: compact A, epilogue scatters via rowIdx masked by cnt.
// BM=128 BN=64 BK=64; 256 threads = 8 waves; wave tile 32x32; 8 WMMA/chunk.
// Software pipeline: global loads for chunk kc+1 are issued before computing
// chunk kc; convert + ds_store happen after the WMMAs, so load latency hides.
// LDS rows padded to 144B -> the 16 lanes' ds_load_b128 tile all 64 banks.
// ---------------------------------------------------------------------------
template<bool BT, int EPI, int GM>
__global__ __launch_bounds__(256) void gemm_kernel(
    const float* __restrict__ A, const float* __restrict__ B,
    float* __restrict__ C, const float* __restrict__ aux, int auxStride,
    const int* __restrict__ rowIdx, const int* __restrict__ cntPtr,
    int M, int N, int K)
{
    constexpr int BM = 128, BN = 64, BK = 64, LDST = 72; // LDST in bf16 elems (144B)
    __shared__ __align__(16) unsigned short lds[2 * (BM + BN) * LDST];

    const int t  = threadIdx.x;
    const int m0 = blockIdx.y * BM;
    const int n0 = blockIdx.x * BN;
    const int nk = K / BK;

    int count = M;
    if (GM != 0) {
        count = *cntPtr;
        if (m0 >= count) return;      // uniform across block; before any barrier
    }

    const int lane = t & 31, wid = t >> 5;
    const int r = lane & 15, h = lane >> 4;
    const int wm = (wid & 3) * 32;      // 4 waves along M
    const int wn = (wid >> 2) * 32;     // 2 waves along N

    // Hoisted global base pointers for the staging pattern.
    int arow = m0 + (t >> 1);
    if (GM == 1) arow = rowIdx[arow];
    const float* aBase = A + (size_t)arow * K + (t & 1) * 32;   // 32 f32 per thread

    const float* bBase;
    if (BT) bBase = B + (size_t)(n0 + (t >> 2)) * K + (t & 3) * 16;   // 16 f32
    else    bBase = B + (size_t)(t >> 2) * N + n0 + (t & 3) * 16;     // 16 f32

    v8f acc[2][2] = {};

    struct SReg { float4 a[8]; float4 b[4]; };

    auto gload = [&](int kc) {
        SReg s;
        const int k0 = kc * BK;
        const float4* sa = reinterpret_cast<const float4*>(aBase + k0);
        #pragma unroll
        for (int i = 0; i < 8; ++i) s.a[i] = sa[i];
        const float4* sb;
        if (BT) sb = reinterpret_cast<const float4*>(bBase + k0);
        else    sb = reinterpret_cast<const float4*>(bBase + (size_t)k0 * N);
        #pragma unroll
        for (int i = 0; i < 4; ++i) s.b[i] = sb[i];
        return s;
    };

    auto gstore = [&](int buf, const SReg& s) {
        unsigned short* la = lds + buf * (BM + BN) * LDST;
        unsigned short* lb = la + BM * LDST;
        // A: 32 contiguous bf16 per thread.
        uint4 d[4];
        #pragma unroll
        for (int i = 0; i < 4; ++i) {
            d[i].x = cvt2(s.a[2*i].x,   s.a[2*i].y);
            d[i].y = cvt2(s.a[2*i].z,   s.a[2*i].w);
            d[i].z = cvt2(s.a[2*i+1].x, s.a[2*i+1].y);
            d[i].w = cvt2(s.a[2*i+1].z, s.a[2*i+1].w);
        }
        uint4* dstA = reinterpret_cast<uint4*>(la + (t >> 1) * LDST + (t & 1) * 32);
        #pragma unroll
        for (int i = 0; i < 4; ++i) dstA[i] = d[i];
        // B
        if (BT) {
            uint4 e0, e1;
            e0.x = cvt2(s.b[0].x, s.b[0].y); e0.y = cvt2(s.b[0].z, s.b[0].w);
            e0.z = cvt2(s.b[1].x, s.b[1].y); e0.w = cvt2(s.b[1].z, s.b[1].w);
            e1.x = cvt2(s.b[2].x, s.b[2].y); e1.y = cvt2(s.b[2].z, s.b[2].w);
            e1.z = cvt2(s.b[3].x, s.b[3].y); e1.w = cvt2(s.b[3].z, s.b[3].w);
            uint4* dstB = reinterpret_cast<uint4*>(lb + (t >> 2) * LDST + (t & 3) * 16);
            dstB[0] = e0;
            dstB[1] = e1;
        } else {
            // B chunk row kr = t>>2, 16 n-values -> transposed scalar stores.
            float fv[16] = { s.b[0].x, s.b[0].y, s.b[0].z, s.b[0].w,
                             s.b[1].x, s.b[1].y, s.b[1].z, s.b[1].w,
                             s.b[2].x, s.b[2].y, s.b[2].z, s.b[2].w,
                             s.b[3].x, s.b[3].y, s.b[3].z, s.b[3].w };
            int kr  = t >> 2;
            int nof = (t & 3) * 16;
            #pragma unroll
            for (int j = 0; j < 16; ++j) lb[(nof + j) * LDST + kr] = cvt1(fv[j]);
        }
    };

    SReg sr = gload(0);
    gstore(0, sr);
    __syncthreads();

    for (int kc = 0; kc < nk; ++kc) {
        const int cur = kc & 1;
        const bool more = kc + 1 < nk;
        if (more) sr = gload(kc + 1);          // issue early; wait only in gstore

        unsigned short* la = lds + cur * (BM + BN) * LDST;
        unsigned short* lb = la + BM * LDST;

        #pragma unroll
        for (int s = 0; s < 2; ++s) {          // two 32-deep K steps per chunk
            Frag a[2], b[2];
            #pragma unroll
            for (int i = 0; i < 2; ++i) {
                const uint4* p = reinterpret_cast<const uint4*>(
                    la + (wm + i*16 + r) * LDST + s * 32 + h * 8);
                a[i].q[0] = p[0];   // K = s*32 + h*8 ..
                a[i].q[1] = p[2];   // K = s*32 + 16 + h*8 ..
            }
            #pragma unroll
            for (int j = 0; j < 2; ++j) {
                const uint4* p = reinterpret_cast<const uint4*>(
                    lb + (wn + j*16 + r) * LDST + s * 32 + h * 8);
                b[j].q[0] = p[0];
                b[j].q[1] = p[2];
            }
            #pragma unroll
            for (int i = 0; i < 2; ++i)
                #pragma unroll
                for (int j = 0; j < 2; ++j)
                    acc[i][j] = __builtin_amdgcn_wmma_f32_16x16x32_bf16(
                        false, a[i].v, false, b[j].v, (short)0, acc[i][j], false, false);
        }

        if (more) gstore(cur ^ 1, sr);
        __syncthreads();
    }

    // Epilogue. C/D layout: vgpr e, lane -> m = e + 8*(lane/16), n = lane%16.
    #pragma unroll
    for (int i = 0; i < 2; ++i)
        #pragma unroll
        for (int e = 0; e < 8; ++e) {
            const int cm = m0 + wm + i*16 + e + 8*h;     // compact row index
            int tok = cm;
            bool wr = true;
            if (GM == 2) {
                wr  = cm < count;
                tok = wr ? rowIdx[cm] : 0;
            }
            #pragma unroll
            for (int j = 0; j < 2; ++j) {
                int n = n0 + wn + j*16 + r;
                float val = acc[i][j][e];
                if (EPI == 2) {
                    if (wr) {
                        size_t idx = (size_t)tok * N + n;
                        C[idx] = C[idx] + aux[(size_t)tok * auxStride] * val;
                    }
                } else {
                    size_t idx = (size_t)cm * N + n;
                    if (EPI == 1) val += aux[idx];
                    if (EPI == 3) val = (val / (1.f + __expf(-val))) * aux[idx];
                    C[idx] = val;
                }
            }
        }
}

// ---------------------------------------------------------------------------
// RMSNorm: one block per row (N = 2880).
// ---------------------------------------------------------------------------
__global__ __launch_bounds__(256) void rmsnorm_kernel(
    const float* __restrict__ x, const float* __restrict__ w, float* __restrict__ y)
{
    __shared__ float red[8];
    const int row = blockIdx.x, t = threadIdx.x;
    const float* xr = x + (size_t)row * 2880;
    float ss = 0.f;
    for (int i = t; i < 2880; i += 256) { float v = xr[i]; ss += v * v; }
    #pragma unroll
    for (int m = 16; m >= 1; m >>= 1) ss += __shfl_xor(ss, m);
    if ((t & 31) == 0) red[t >> 5] = ss;
    __syncthreads();
    float tot = 0.f;
    #pragma unroll
    for (int i = 0; i < 8; ++i) tot += red[i];
    const float rinv = rsqrtf(tot * (1.0f / 2880.0f) + 1e-5f);
    float* yr = y + (size_t)row * 2880;
    for (int i = t; i < 2880; i += 256) yr[i] = xr[i] * rinv * w[i];
}

// ---------------------------------------------------------------------------
// RoPE + bf16 pack. Reads f32 q[1024,32,64] / k[1024,8,64], writes bf16
// Qb (with 1/sqrt(64) softmax scale folded in) and bf16 Kb.
// ---------------------------------------------------------------------------
__global__ __launch_bounds__(256) void rope_pack_kernel(
    const float* __restrict__ q, const float* __restrict__ k,
    const int* __restrict__ pos,
    unsigned short* __restrict__ Qb, unsigned short* __restrict__ Kb)
{
    int tid  = blockIdx.x * 256 + threadIdx.x;      // 1024*40*32 total
    int pair = tid & 31;
    int hd   = (tid >> 5) % 40;
    int tok  = tid / (32 * 40);
    if (tok >= 1024) return;
    float p    = (float)pos[tok];
    float invf = __powf(150000.0f, -(float)(2 * pair) * (1.0f / 64.0f));
    float s, c;
    __sincosf(p * invf, &s, &c);
    if (hd < 32) {
        const float* base = q + (size_t)tok * 2048 + hd * 64;
        float x1 = base[pair], x2 = base[pair + 32];
        unsigned short* ob = Qb + (size_t)tok * 2048 + hd * 64;
        ob[pair]      = cvt1((x1 * c - x2 * s) * 0.125f);
        ob[pair + 32] = cvt1((x2 * c + x1 * s) * 0.125f);
    } else {
        const float* base = k + (size_t)tok * 512 + (hd - 32) * 64;
        float x1 = base[pair], x2 = base[pair + 32];
        unsigned short* ob = Kb + (size_t)tok * 512 + (hd - 32) * 64;
        ob[pair]      = cvt1(x1 * c - x2 * s);
        ob[pair + 32] = cvt1(x2 * c + x1 * s);
    }
}

// ---------------------------------------------------------------------------
// Pack V [1024 tok][512] f32 -> transposed bf16 Vt [512][1024 tok].
// ---------------------------------------------------------------------------
__global__ __launch_bounds__(256) void vpack_kernel(
    const float* __restrict__ V, unsigned short* __restrict__ Vt)
{
    int idx = blockIdx.x * 256 + threadIdx.x;       // 512*1024 total
    int tok = idx & 1023, d = idx >> 10;
    Vt[(size_t)d * 1024 + tok] = cvt1(V[(size_t)tok * 512 + d]);
}

// ---------------------------------------------------------------------------
// Flash attention (causal, GQA 32q/8kv heads, D=64). One wave per
// (head, 16-row q block); keys processed 32 at a time; bf16 WMMA throughout.
// ---------------------------------------------------------------------------
__global__ __launch_bounds__(256) void attn_kernel(
    const unsigned short* __restrict__ Qb, const unsigned short* __restrict__ Kb,
    const unsigned short* __restrict__ Vt, float* __restrict__ O)
{
    __shared__ __align__(16) unsigned short pl[8 * 16 * 40]; // per-wave P tile
    const int t = threadIdx.x, lane = t & 31, wid = t >> 5;
    const int r = lane & 15, h = lane >> 4;
    const int gw   = blockIdx.x * 8 + wid;
    const int head = gw >> 6;
    const int qb   = gw & 63;
    const int kvh  = head >> 2;
    unsigned short* pw = pl + wid * 16 * 40;

    Frag aq[2];
    #pragma unroll
    for (int c = 0; c < 2; ++c) {
        const uint4* src = reinterpret_cast<const uint4*>(
            Qb + (size_t)(qb * 16 + r) * 2048 + head * 64 + c * 32 + h * 8);
        aq[c].q[0] = src[0];
        aq[c].q[1] = src[2];
    }

    v8f o[4] = {};
    float rmax[8], rsum[8];
    #pragma unroll
    for (int e = 0; e < 8; ++e) { rmax[e] = -3.0e38f; rsum[e] = 0.f; }

    const int kbmax = qb >> 1;
    for (int kb = 0; kb <= kbmax; ++kb) {
        v8f st[2];
        #pragma unroll
        for (int jt = 0; jt < 2; ++jt) {
            const int j = kb * 32 + jt * 16;
            Frag bk[2];
            #pragma unroll
            for (int c = 0; c < 2; ++c) {
                const uint4* src = reinterpret_cast<const uint4*>(
                    Kb + (size_t)(j + r) * 512 + kvh * 64 + c * 32 + h * 8);
                bk[c].q[0] = src[0];
                bk[c].q[1] = src[2];
            }
            v8f sc = {};
            sc = __builtin_amdgcn_wmma_f32_16x16x32_bf16(false, aq[0].v, false, bk[0].v,
                                                         (short)0, sc, false, false);
            sc = __builtin_amdgcn_wmma_f32_16x16x32_bf16(false, aq[1].v, false, bk[1].v,
                                                         (short)0, sc, false, false);
            #pragma unroll
            for (int e = 0; e < 8; ++e) {
                int qt = qb * 16 + e + 8 * h;
                int kt = j + r;
                if (kt > qt) sc[e] = -3.0e38f;
            }
            st[jt] = sc;
        }
        float alpha[8];
        #pragma unroll
        for (int e = 0; e < 8; ++e) {
            float v = fmaxf(st[0][e], st[1][e]);
            #pragma unroll
            for (int m = 8; m >= 1; m >>= 1) v = fmaxf(v, __shfl_xor(v, m));
            float mn = fmaxf(rmax[e], v);
            alpha[e] = __expf(rmax[e] - mn);
            rmax[e]  = mn;
        }
        float lsum[8];
        #pragma unroll
        for (int e = 0; e < 8; ++e) lsum[e] = 0.f;
        #pragma unroll
        for (int jt = 0; jt < 2; ++jt)
            #pragma unroll
            for (int e = 0; e < 8; ++e) {
                float p = __expf(st[jt][e] - rmax[e]);
                lsum[e] += p;
                pw[(e + 8 * h) * 40 + jt * 16 + r] = cvt1(p);
            }
        #pragma unroll
        for (int e = 0; e < 8; ++e) {
            float v = lsum[e];
            #pragma unroll
            for (int m = 8; m >= 1; m >>= 1) v += __shfl_xor(v, m);
            rsum[e] = rsum[e] * alpha[e] + v;
        }
        #pragma unroll
        for (int tn = 0; tn < 4; ++tn)
            #pragma unroll
            for (int e = 0; e < 8; ++e) o[tn][e] *= alpha[e];

        asm volatile("s_wait_dscnt 0x0" ::: "memory"); // pl writes visible (same wave)

        Frag pa;
        {
            const uint4* p = reinterpret_cast<const uint4*>(pw + r * 40 + h * 8);
            pa.q[0] = p[0];
            pa.q[1] = p[2];
        }
        #pragma unroll
        for (int tn = 0; tn < 4; ++tn) {
            Frag bv;
            const uint4* src = reinterpret_cast<const uint4*>(
                Vt + (size_t)(kvh * 64 + tn * 16 + r) * 1024 + kb * 32 + h * 8);
            bv.q[0] = src[0];
            bv.q[1] = src[2];
            o[tn] = __builtin_amdgcn_wmma_f32_16x16x32_bf16(false, pa.v, false, bv.v,
                                                            (short)0, o[tn], false, false);
        }
    }

    #pragma unroll
    for (int e = 0; e < 8; ++e) rsum[e] = 1.0f / rsum[e];
    #pragma unroll
    for (int tn = 0; tn < 4; ++tn)
        #pragma unroll
        for (int e = 0; e < 8; ++e)
            O[(size_t)(qb * 16 + e + 8 * h) * 2048 + head * 64 + tn * 16 + r] =
                o[tn][e] * rsum[e];
}

// ---------------------------------------------------------------------------
// Router: logits -> softmax -> top-4 routing weights (zeros elsewhere).
// ---------------------------------------------------------------------------
__global__ __launch_bounds__(256) void router_kernel(
    const float* __restrict__ x, const float* __restrict__ rw, float* __restrict__ wexp)
{
    __shared__ float sm[8][8];
    const int row = blockIdx.x, t = threadIdx.x;
    const float* xr = x + (size_t)row * 2880;
    float l8[8];
    #pragma unroll
    for (int e = 0; e < 8; ++e) l8[e] = 0.f;
    for (int i = t; i < 2880; i += 256) {
        float xv = xr[i];
        #pragma unroll
        for (int e = 0; e < 8; ++e) l8[e] += xv * rw[e * 2880 + i];
    }
    #pragma unroll
    for (int e = 0; e < 8; ++e) {
        float v = l8[e];
        #pragma unroll
        for (int m = 16; m >= 1; m >>= 1) v += __shfl_xor(v, m);
        l8[e] = v;
    }
    if ((t & 31) == 0)
        for (int e = 0; e < 8; ++e) sm[t >> 5][e] = l8[e];
    __syncthreads();
    if (t == 0) {
        float logits[8];
        for (int e = 0; e < 8; ++e) {
            float v = 0.f;
            for (int w2 = 0; w2 < 8; ++w2) v += sm[w2][e];
            logits[e] = v;
        }
        float mx = logits[0];
        for (int e = 1; e < 8; ++e) mx = fmaxf(mx, logits[e]);
        float den = 0.f, pr[8];
        for (int e = 0; e < 8; ++e) { pr[e] = __expf(logits[e] - mx); den += pr[e]; }
        float out[8];
        bool used[8];
        for (int e = 0; e < 8; ++e) { pr[e] /= den; out[e] = 0.f; used[e] = false; }
        for (int sel = 0; sel < 4; ++sel) {
            int bi = 0; float bv = -1.f;
            for (int e = 0; e < 8; ++e)
                if (!used[e] && pr[e] > bv) { bv = pr[e]; bi = e; }
            used[bi] = true; out[bi] = bv;
        }
        for (int e = 0; e < 8; ++e) wexp[row * 8 + e] = out[e];
    }
}

// ---------------------------------------------------------------------------
// Deterministic per-expert token compaction: one wave per expert, in-order
// ballot/popc prefix over 32-token chunks.
// ---------------------------------------------------------------------------
__global__ __launch_bounds__(32) void expert_gather_kernel(
    const float* __restrict__ wexp, int* __restrict__ list, int* __restrict__ cnt)
{
    const int e = blockIdx.x;       // 8 experts
    const int lane = threadIdx.x;   // 32 lanes
    int* le = list + e * 1024;
    for (int i = lane; i < 1024; i += 32) le[i] = 0;   // pad with token 0
    int base = 0;
    for (int i0 = 0; i0 < 1024; i0 += 32) {
        int tok = i0 + lane;
        bool f = wexp[tok * 8 + e] > 0.f;
        unsigned m32 = (unsigned)__ballot(f);
        int below = __popc(m32 & ((1u << lane) - 1u));
        if (f) le[base + below] = tok;
        base += __popc(m32);
    }
    if (lane == 0) cnt[e] = base;
}

__global__ __launch_bounds__(256) void copy_kernel(
    const float* __restrict__ a, float* __restrict__ b, int n)
{
    int i = blockIdx.x * 256 + threadIdx.x;
    if (i < n) b[i] = a[i];
}

// ---------------------------------------------------------------------------
// Launch
// ---------------------------------------------------------------------------
extern "C" void kernel_launch(void* const* d_in, const int* in_sizes, int n_in,
                              void* d_out, int out_size, void* d_ws, size_t ws_size,
                              hipStream_t stream)
{
    (void)in_sizes; (void)n_in; (void)out_size; (void)ws_size;
    const float* hidden = (const float*)d_in[0];
    const int*   pos    = (const int*)  d_in[1];
    const float* ln1    = (const float*)d_in[2];
    const float* ln2    = (const float*)d_in[3];
    const float* wq     = (const float*)d_in[4];
    const float* wk     = (const float*)d_in[5];
    const float* wv     = (const float*)d_in[6];
    const float* wo     = (const float*)d_in[7];
    const float* rw     = (const float*)d_in[8];
    const float* wg     = (const float*)d_in[9];
    const float* wu     = (const float*)d_in[10];
    const float* wd     = (const float*)d_in[11];
    float* out = (float*)d_out;

    float* ws  = (float*)d_ws;
    float* h1n = ws;  ws += (size_t)1024 * 2880;
    float* qf  = ws;  ws += (size_t)1024 * 2048;
    float* kf  = ws;  ws += (size_t)1024 * 512;
    float* vf  = ws;  ws += (size_t)1024 * 512;
    float* at  = ws;  ws += (size_t)1024 * 2048;
    float* hrs = ws;  ws += (size_t)1024 * 2880;
    float* h2n = ws;  ws += (size_t)1024 * 2880;
    float* we  = ws;  ws += (size_t)1024 * 8;
    float* gb  = ws;  ws += (size_t)1024 * 2880;
    float* ub  = ws;  ws += (size_t)1024 * 2880;
    unsigned short* Qb = (unsigned short*)ws;  ws += (size_t)1024 * 2048 / 2;
    unsigned short* Kb = (unsigned short*)ws;  ws += (size_t)1024 * 512 / 2;
    unsigned short* Vt = (unsigned short*)ws;  ws += (size_t)1024 * 512 / 2;
    int* elist = (int*)ws;  ws += (size_t)8 * 1024;
    int* ecnt  = elist + 8 * 1024;

    dim3 blk(256);

    rmsnorm_kernel<<<1024, blk, 0, stream>>>(hidden, ln1, h1n);
    gemm_kernel<true, 0, 0><<<dim3(32, 8), blk, 0, stream>>>(h1n, wq, qf, nullptr, 0, nullptr, nullptr, 1024, 2048, 2880);
    gemm_kernel<true, 0, 0><<<dim3(8, 8),  blk, 0, stream>>>(h1n, wk, kf, nullptr, 0, nullptr, nullptr, 1024, 512, 2880);
    gemm_kernel<true, 0, 0><<<dim3(8, 8),  blk, 0, stream>>>(h1n, wv, vf, nullptr, 0, nullptr, nullptr, 1024, 512, 2880);
    rope_pack_kernel<<<5120, blk, 0, stream>>>(qf, kf, pos, Qb, Kb);
    vpack_kernel<<<2048, blk, 0, stream>>>(vf, Vt);
    attn_kernel<<<256, blk, 0, stream>>>(Qb, Kb, Vt, at);
    gemm_kernel<true, 1, 0><<<dim3(45, 8), blk, 0, stream>>>(at, wo, hrs, hidden, 0, nullptr, nullptr, 1024, 2880, 2048);
    rmsnorm_kernel<<<1024, blk, 0, stream>>>(hrs, ln2, h2n);
    router_kernel<<<1024, blk, 0, stream>>>(h2n, rw, we);
    expert_gather_kernel<<<8, 32, 0, stream>>>(we, elist, ecnt);
    copy_kernel<<<(1024 * 2880 + 255) / 256, blk, 0, stream>>>(hrs, out, 1024 * 2880);

    for (int e = 0; e < 8; ++e) {
        const float* wge = wg + (size_t)e * 2880 * 2880;
        const float* wue = wu + (size_t)e * 2880 * 2880;
        const float* wde = wd + (size_t)e * 2880 * 2880;
        const int* lst = elist + e * 1024;
        const int* cp  = ecnt + e;
        // up projection (gathered tokens) -> ub (compact)
        gemm_kernel<false, 0, 1><<<dim3(45, 8), blk, 0, stream>>>(h2n, wue, ub, nullptr, 0, lst, cp, 1024, 2880, 2880);
        // gate projection with fused SiLU * up -> gb (compact)
        gemm_kernel<false, 3, 1><<<dim3(45, 8), blk, 0, stream>>>(h2n, wge, gb, ub, 0, lst, cp, 1024, 2880, 2880);
        // down projection, routed scatter-accumulate into out
        gemm_kernel<false, 2, 2><<<dim3(45, 8), blk, 0, stream>>>(gb, wde, out, we + e, 8, lst, cp, 1024, 2880, 2880);
    }
}
